// GCNN_25872882991625
// MI455X (gfx1250) — matrix operand order repaired and checked by reading, not compiled, not run on hardware
//
#include <hip/hip_runtime.h>
#include <stdint.h>

#define D_DIM 1024
#define OUTD  128
#define NGRAPH 64

typedef __attribute__((ext_vector_type(16))) __bf16    v16bf;
typedef __attribute__((ext_vector_type(8)))  float     v8f;
typedef __attribute__((ext_vector_type(8)))  uint32_t  v8u;

__device__ inline float lrelu(float x) { return x >= 0.0f ? x : 0.01f * x; }

// round-to-nearest-even fp32 -> bf16 bits
__device__ inline uint16_t f2bf(float f) {
    uint32_t u = __float_as_uint(f);
    uint32_t r = (u + 0x7FFFu + ((u >> 16) & 1u)) >> 16;
    return (uint16_t)r;
}
__device__ inline uint32_t pack2bf(float a, float b) {
    return (uint32_t)f2bf(a) | ((uint32_t)f2bf(b) << 16);
}

__device__ inline v16bf load_frag_lds(const char* p) {
    v8u t = *(const v8u*)p;                  // 32B/lane -> 2x ds_load_b128
    return __builtin_bit_cast(v16bf, t);
}

// async global -> LDS copy, 16B per lane, tracked by ASYNCcnt
__device__ inline void async_copy_b128(uint32_t lds_off, const void* gaddr) {
    asm volatile("global_load_async_to_lds_b128 %0, %1, off"
                 :: "v"(lds_off), "v"(gaddr)
                 : "memory");
}
__device__ inline void wait_async0() {
    asm volatile("s_wait_asynccnt 0x0" ::: "memory");
}

// ---------------------------------------------------------------------------
// fill
// ---------------------------------------------------------------------------
__global__ void k_fill(float* __restrict__ p, float val, unsigned n) {
    unsigned i = blockIdx.x * 256u + threadIdx.x;
    if (i < n) p[i] = val;
}

// ---------------------------------------------------------------------------
// A pre-swizzle: x[N x 1024] fp32 -> bf16 fragments, CDNA5 16x32 A layout.
// Frag (mt,kt) = 1KB; lane owns 32B: VGPR v packs (k0,k0+1) where
// k0 = (v<4?0:16) + 2*(v&3) + (lane>=16 ? 8 : 0), row m = mt*16 + lane%16.
// Rows m >= nRows are zero-filled (padding to the 128-row block grid).
// ---------------------------------------------------------------------------
__global__ void __launch_bounds__(256) k_convert_A(const float* __restrict__ x,
                                                   uint32_t* __restrict__ out,
                                                   int nRows) {
    int frag   = blockIdx.x;            // mt*(D/32) + kt
    int within = threadIdx.x;           // 256 dwords per frag
    int lane = within >> 3, v = within & 7;
    int mt = frag >> 5;                 // D/32 == 32
    int kt = frag & 31;
    int m  = mt * 16 + (lane & 15);
    int hi = lane >> 4;
    int k0 = kt * 32 + ((v < 4) ? 0 : 16) + 2 * (v & 3) + (hi ? 8 : 0);
    float a = 0.0f, b = 0.0f;
    if (m < nRows) {
        a = x[(size_t)m * D_DIM + k0];
        b = x[(size_t)m * D_DIM + k0 + 1];
    }
    out[(size_t)frag * 256 + within] = pack2bf(a, b);
}

// ---------------------------------------------------------------------------
// B pre-swizzle: W[1024 x 1024] fp32 (h = x@W, W[k][n]) -> bf16 32x16 B frags.
// Frag (kt,nt): lane = n%16 + 16*hi within tile, VGPR v packs (k0,k0+1) with
// k0 = (v<4?0:16) + 2*(v&3) + (hi ? 8 : 0).  Frags (kt, nt=0..63) are contiguous.
// ---------------------------------------------------------------------------
__global__ void __launch_bounds__(256) k_convert_B(const float* __restrict__ W,
                                                   uint32_t* __restrict__ out) {
    int frag   = blockIdx.x;            // kt*(D/16) + nt
    int within = threadIdx.x;
    int lane = within >> 3, v = within & 7;
    int kt = frag >> 6;                 // D/16 == 64
    int nt = frag & 63;
    int n  = nt * 16 + (lane & 15);
    int hi = lane >> 4;
    int k0 = kt * 32 + ((v < 4) ? 0 : 16) + 2 * (v & 3) + (hi ? 8 : 0);
    out[(size_t)frag * 256 + within] =
        pack2bf(W[(size_t)k0 * D_DIM + n], W[(size_t)(k0 + 1) * D_DIM + n]);
}

// ---------------------------------------------------------------------------
// GEMM: H[N x 1024] = A @ W, bf16 WMMA, f32 accum.
// Block = 8 waves = 128(M) x 128(N) tile. Per K-step: async-DMA the 8 A frags
// + 8 B frags (16KB) into LDS (double-buffered), then 8 WMMAs per wave.
// ---------------------------------------------------------------------------
#define TILE_BYTES 16384     // 8 A frags + 8 B frags per K-step

__global__ void __launch_bounds__(256) k_gemm(const uint32_t* __restrict__ A,
                                              const uint32_t* __restrict__ Bm,
                                              float* __restrict__ H, int nRows) {
    __shared__ char smem[2 * TILE_BYTES];

    const int mBlk = blockIdx.x;        // 128-row block
    const int nBlk = blockIdx.y;        // 128-col block
    const int t    = threadIdx.x;
    const int w    = t >> 5;            // wave in block: M subtile
    const int lane = t & 31;
    const uint32_t base = (uint32_t)(uintptr_t)&smem[0];

    // cooperative async tile loader for K-step kt into buffer buf
    auto issue_tile = [&](int buf, int kt) {
        uint32_t dst = base + (uint32_t)buf * TILE_BYTES;
        // A: 8 frags of 1KB, frag f at global frag index (mBlk*8+f)*32 + kt
        {
            int c = t & 63;             // 16B chunk within frag
            int f0 = t >> 6;            // 4 frags per round
            #pragma unroll
            for (int r = 0; r < 2; ++r) {
                int f = r * 4 + f0;
                const uint32_t* g =
                    A + (size_t)((mBlk * 8 + f) * 32 + kt) * 256 + c * 4;
                async_copy_b128(dst + f * 1024 + c * 16, g);
            }
        }
        // B: 8 contiguous frags (8KB) at frag index kt*64 + nBlk*8
        {
            const uint32_t* gB = Bm + (size_t)(kt * 64 + nBlk * 8) * 256;
            #pragma unroll
            for (int r = 0; r < 2; ++r) {
                int ci = r * 256 + t;   // 16B chunk index in 8KB
                async_copy_b128(dst + 8192 + ci * 16, gB + ci * 4);
            }
        }
    };

    v8f acc[8];
    #pragma unroll
    for (int i = 0; i < 8; ++i) acc[i] = (v8f)0.0f;

    issue_tile(0, 0);
    int buf = 0;
    for (int kt = 0; kt < 32; ++kt) {
        wait_async0();                  // my copies (incl. this buf) landed
        __syncthreads();                // everyone's copies landed; prev reads done
        if (kt + 1 < 32) issue_tile(buf ^ 1, kt + 1);   // overlap DMA with WMMA

        const char* tileA = smem + buf * TILE_BYTES + lane * 32;
        const char* tileB = smem + buf * TILE_BYTES + 8192 + lane * 32;
        v16bf a = load_frag_lds(tileA + w * 1024);
        #pragma unroll
        for (int nt = 0; nt < 8; ++nt) {
            v16bf b = load_frag_lds(tileB + nt * 1024);
            acc[nt] = __builtin_amdgcn_wmma_f32_16x16x32_bf16(
                false, a, false, b, (short)0, acc[nt], false, false);
        }
        buf ^= 1;
    }

    // C/D layout: VGPR v -> M = v + (lane>=16 ? 8 : 0), N = lane%16
    int n0    = nBlk * 128 + (lane & 15);
    int mBase = (mBlk * 8 + w) * 16 + ((lane >> 4) ? 8 : 0);
    #pragma unroll
    for (int nt = 0; nt < 8; ++nt) {
        #pragma unroll
        for (int v = 0; v < 8; ++v) {
            int m = mBase + v;
            if (m < nRows) H[(size_t)m * D_DIM + n0 + nt * 16] = acc[nt][v];
        }
    }
}

// ---------------------------------------------------------------------------
// degree: deg[dst] += 1   (deg pre-filled with 1.0 = self loop)
// ---------------------------------------------------------------------------
__global__ void k_degree(const int* __restrict__ ei, float* __restrict__ deg, int E) {
    int e = blockIdx.x * 256 + threadIdx.x;
    if (e < E) atomicAdd(&deg[ei[E + e]], 1.0f);
}

// ---------------------------------------------------------------------------
// edge scatter: agg[dst] += h[src] * rsqrt(deg[src])*rsqrt(deg[dst])
// one block per edge, 256 threads x float4 (full 1024-wide row), L2-resident.
// ---------------------------------------------------------------------------
__global__ void __launch_bounds__(256) k_edge_agg(const int* __restrict__ ei,
                                                  const float* __restrict__ deg,
                                                  const float* __restrict__ H,
                                                  float* __restrict__ agg, int E) {
    int e   = blockIdx.x;
    int src = ei[e];
    int dst = ei[E + e];
    float norm = rsqrtf(deg[src]) * rsqrtf(deg[dst]);
    const float4* hs = (const float4*)(H + (size_t)src * D_DIM);
    float* ad = agg + (size_t)dst * D_DIM;
    float4 h = hs[threadIdx.x];
    int o = threadIdx.x * 4;
    atomicAdd(ad + o + 0, h.x * norm);
    atomicAdd(ad + o + 1, h.y * norm);
    atomicAdd(ad + o + 2, h.z * norm);
    atomicAdd(ad + o + 3, h.w * norm);
}

// ---------------------------------------------------------------------------
// finalize node + pool: val = lrelu(agg + h/deg + bias); pool[batch[n]] += val
// ---------------------------------------------------------------------------
__global__ void __launch_bounds__(256) k_finalize_pool(const float* __restrict__ agg,
                                                       const float* __restrict__ H,
                                                       const float* __restrict__ deg,
                                                       const float* __restrict__ bias,
                                                       const int* __restrict__ batch,
                                                       float* __restrict__ pool) {
    int n = blockIdx.x;
    int t = threadIdx.x;
    float invd = 1.0f / deg[n];
    int g = batch[n];
    float4 a  = ((const float4*)(agg + (size_t)n * D_DIM))[t];
    float4 h  = ((const float4*)(H   + (size_t)n * D_DIM))[t];
    float4 bb = ((const float4*)bias)[t];
    float* pg = pool + (size_t)g * D_DIM + t * 4;
    atomicAdd(pg + 0, lrelu(a.x + h.x * invd + bb.x));
    atomicAdd(pg + 1, lrelu(a.y + h.y * invd + bb.y));
    atomicAdd(pg + 2, lrelu(a.z + h.z * invd + bb.z));
    atomicAdd(pg + 3, lrelu(a.w + h.w * invd + bb.w));
}

__global__ void k_count(const int* __restrict__ batch, float* __restrict__ cnt, int n) {
    int i = blockIdx.x * 256 + threadIdx.x;
    if (i < n) atomicAdd(&cnt[batch[i]], 1.0f);
}

// ---------------------------------------------------------------------------
// FC head: out[g][o] = lrelu( mean(pool[g]) . fcW[:,o] + fcb[o] )
// ---------------------------------------------------------------------------
__global__ void __launch_bounds__(OUTD) k_fc(const float* __restrict__ pool,
                                             const float* __restrict__ cnt,
                                             const float* __restrict__ fcW,
                                             const float* __restrict__ fcb,
                                             float* __restrict__ out) {
    __shared__ float mean[D_DIM];
    int g = blockIdx.x;
    int t = threadIdx.x;
    float inv = 1.0f / fmaxf(cnt[g], 1.0f);
    for (int d = t; d < D_DIM; d += OUTD)
        mean[d] = pool[(size_t)g * D_DIM + d] * inv;
    __syncthreads();
    float acc = fcb[t];
    for (int d = 0; d < D_DIM; ++d)
        acc = fmaf(mean[d], fcW[(size_t)d * OUTD + t], acc);
    out[(size_t)g * OUTD + t] = lrelu(acc);
}

// ---------------------------------------------------------------------------
// final: out[g] = [fc1out | fc2out] . final_W + final_b
// ---------------------------------------------------------------------------
__global__ void k_final(const float* __restrict__ f1, const float* __restrict__ f2,
                        const float* __restrict__ W, const float* __restrict__ b,
                        float* __restrict__ out) {
    int g = threadIdx.x;
    if (g < NGRAPH) {
        float acc = b[0];
        for (int j = 0; j < OUTD; ++j) acc = fmaf(f1[g * OUTD + j], W[j], acc);
        for (int j = 0; j < OUTD; ++j) acc = fmaf(f2[g * OUTD + j], W[OUTD + j], acc);
        out[g] = acc;
    }
}

// ---------------------------------------------------------------------------
extern "C" void kernel_launch(void* const* d_in, const int* in_sizes, int n_in,
                              void* d_out, int out_size, void* d_ws, size_t ws_size,
                              hipStream_t stream) {
    const float* x1   = (const float*)d_in[0];
    const int*   e1   = (const int*)  d_in[1];
    const int*   bat1 = (const int*)  d_in[2];
    const float* x2   = (const float*)d_in[3];
    const int*   e2   = (const int*)  d_in[4];
    const int*   bat2 = (const int*)  d_in[5];
    const float* W1   = (const float*)d_in[6];
    const float* b1   = (const float*)d_in[7];
    const float* fc1W = (const float*)d_in[8];
    const float* fc1b = (const float*)d_in[9];
    const float* W2   = (const float*)d_in[10];
    const float* b2   = (const float*)d_in[11];
    const float* fc2W = (const float*)d_in[12];
    const float* fc2b = (const float*)d_in[13];
    const float* fW   = (const float*)d_in[14];
    const float* fb   = (const float*)d_in[15];

    const int N   = in_sizes[0] / D_DIM;     // 10000
    const int E   = in_sizes[1] / 2;         // 160000
    const int MB  = (N + 127) / 128;         // 79 M-blocks of 128 rows
    const int MTP = MB * 8;                  // padded row tiles (632)

    // workspace carve (big buffers reused across the two sequential branches)
    char* ws = (char*)d_ws;
    size_t off = 0;
    auto carve = [&](size_t bytes) -> void* {
        off = (off + 255) & ~(size_t)255;
        void* p = ws + off;
        off += bytes;
        return p;
    };
    uint32_t* A_sw  = (uint32_t*)carve((size_t)MTP * 32 * 1024);    // bf16 A frags
    uint32_t* B_sw  = (uint32_t*)carve((size_t)2048 * 1024);        // bf16 B frags
    float*    H     = (float*)carve((size_t)N * D_DIM * 4);
    float*    AGG   = (float*)carve((size_t)N * D_DIM * 4);
    float*    DEG   = (float*)carve((size_t)N * 4);
    float*    POOL1 = (float*)carve((size_t)NGRAPH * D_DIM * 4);
    float*    POOL2 = (float*)carve((size_t)NGRAPH * D_DIM * 4);
    float*    CNT1  = (float*)carve((size_t)NGRAPH * 4);
    float*    CNT2  = (float*)carve((size_t)NGRAPH * 4);
    float*    FC1O  = (float*)carve((size_t)NGRAPH * OUTD * 4);
    float*    FC2O  = (float*)carve((size_t)NGRAPH * OUTD * 4);

    unsigned ND = (unsigned)N * D_DIM;

    auto run_branch = [&](const float* x, const int* ei, const int* bat,
                          const float* W, const float* bias,
                          const float* fcW, const float* fcb,
                          float* pool, float* cnt, float* fco) {
        // init accumulators
        k_fill<<<(N + 255) / 256, 256, 0, stream>>>(DEG, 1.0f, (unsigned)N);
        k_fill<<<(ND + 255) / 256, 256, 0, stream>>>(AGG, 0.0f, ND);
        k_fill<<<(NGRAPH * D_DIM + 255) / 256, 256, 0, stream>>>(pool, 0.0f, NGRAPH * D_DIM);
        k_fill<<<1, 256, 0, stream>>>(cnt, 0.0f, NGRAPH);
        // degrees
        k_degree<<<(E + 255) / 256, 256, 0, stream>>>(ei, DEG, E);
        // bf16 pre-swizzle
        k_convert_A<<<MTP * 32, 256, 0, stream>>>(x, A_sw, N);
        k_convert_B<<<2048, 256, 0, stream>>>(W, B_sw);
        // H = x @ W via async-LDS double-buffered WMMA
        dim3 ggrid(MB, 8);
        k_gemm<<<ggrid, 256, 0, stream>>>(A_sw, B_sw, H, N);
        // edge scatter
        k_edge_agg<<<E, 256, 0, stream>>>(ei, DEG, H, AGG, E);
        // finalize + pool
        k_finalize_pool<<<N, 256, 0, stream>>>(AGG, H, DEG, bias, bat, pool);
        k_count<<<(N + 255) / 256, 256, 0, stream>>>(bat, cnt, N);
        // FC head
        k_fc<<<NGRAPH, OUTD, 0, stream>>>(pool, cnt, fcW, fcb, fco);
    };

    run_branch(x1, e1, bat1, W1, b1, fc1W, fc1b, POOL1, CNT1, FC1O);
    run_branch(x2, e2, bat2, W2, b2, fc2W, fc2b, POOL2, CNT2, FC2O);

    k_final<<<1, 64, 0, stream>>>(FC1O, FC2O, fW, fb, (float*)d_out);
}
